// PDFA_18597208391788
// MI455X (gfx1250) — matrix operand order repaired and checked by reading, not compiled, not run on hardware
//
#include <hip/hip_runtime.h>

// Problem constants (match reference)
#define N_SEQ 4096
#define L_T   512
#define C_SYM 16
#define S_ST  32

typedef __attribute__((ext_vector_type(16))) _Float16 v16h;
typedef __attribute__((ext_vector_type(2)))  _Float16 v2h;
typedef __attribute__((ext_vector_type(8)))  float    v8f;
typedef __attribute__((ext_vector_type(4)))  unsigned int v4u;

// Workspace layout (bytes)
//   [0, 32768)        : WMMA A-fragments (f16): 32 frags x 32 lanes x 16 halves
//   [32768, 32896)    : p0 (softmax of initial logits), f32[32]
//   [32896, 33024)    : sigmoid(accepting logits), f32[32]
//   [65536, +64MB)    : x16 = exp(log_input) as f16, relaid to (L, N, C)
#define WS_A  0
#define WS_P0 32768
#define WS_ACC 32896
#define WS_X  65536

// ---------------- Preprocess 1: exp(log_input_probs) -> f16, (N,L,C) -> (L,N,C)
__global__ void prep_x(const float* __restrict__ in, unsigned short* __restrict__ x16) {
  int idx = blockIdx.x * 256 + threadIdx.x;            // 0 .. N*L*C-1
  int b = idx >> 13;                                   // / (L*C) = 8192
  int r = idx & 8191;
  int t = r >> 4;
  int c = r & 15;
  _Float16 h = (_Float16)__expf(in[idx]);
  x16[(size_t)t * (N_SEQ * C_SYM) + (size_t)b * C_SYM + c] =
      __builtin_bit_cast(unsigned short, h);
}

// ---------------- Preprocess 2: softmax(T, axis=2) -> f16, packed in WMMA A-fragment layout
// A[m, k] = T[p, c, m], k = p*C + c.  One block per k (512), one thread per n (32).
__global__ void prep_T(const float* __restrict__ logitT, unsigned short* __restrict__ wsA) {
  int k = blockIdx.x;        // p*C + c
  int n = threadIdx.x;       // destination state = matrix row m
  float e = __expf(logitT[k * S_ST + n]);
  float s = e;
  for (int o = 16; o > 0; o >>= 1) s += __shfl_xor(s, o, 32);
  float v = e / s;

  int kc = k >> 5;                       // K-chunk (K=32 per WMMA)
  int kl = k & 31;                       // local K
  int laneHi = (kl >> 4) & 1;            // which lane half holds this K
  int h = kl & 15;  // half index 0..15 within lane
  int mt = n >> 4;                       // M-tile
  int lm = n & 15;                       // row within tile -> lane % 16
  int lane = laneHi * 16 + lm;
  int frag = mt * 16 + kc;
  _Float16 hv = (_Float16)v;
  wsA[(size_t)(frag * 32 + lane) * 16 + h] = __builtin_bit_cast(unsigned short, hv);
}

// ---------------- Preprocess 3: p0 = softmax(init logits); accw = sigmoid(accept logits)
__global__ void prep_misc(const float* __restrict__ li, const float* __restrict__ la,
                          float* __restrict__ p0, float* __restrict__ accw) {
  int lane = threadIdx.x;  // 32 threads
  float e = __expf(li[lane]);
  float s = e;
  for (int o = 16; o > 0; o >>= 1) s += __shfl_xor(s, o, 32);
  p0[lane] = e / s;
  accw[lane] = 1.0f / (1.0f + __expf(-la[lane]));
}

// ---------------- Main scan kernel: one wave per 16-batch tile, 4 waves per block
__global__ __launch_bounds__(128, 1) void pdfa_main(
    const unsigned short* __restrict__ x16, const unsigned short* __restrict__ wsA,
    const float* __restrict__ p0, const float* __restrict__ accw,
    float* __restrict__ out) {
  __shared__ v16h shA[32 * 32];                 // 32 KB: all 32 A-fragments
  __shared__ __align__(32) float shP[4][16 * 32];  // per-wave state p[b][s], 2 KB each

  int tid = threadIdx.x;

  // Stage A-fragments into LDS (cooperative, 32 KB / 128 threads)
  {
    const v4u* src = (const v4u*)wsA;
    v4u* dst = (v4u*)shA;
#pragma unroll
    for (int i = 0; i < 16; ++i) dst[tid + i * 128] = src[tid + i * 128];
  }
  __syncthreads();

  int wave = tid >> 5;
  int lane = tid & 31;
  int bl = lane & 15;          // batch within tile
  int laneHi = lane >> 4;      // K-half / state-half selector
  int tile = blockIdx.x * 4 + wave;
  int batch = tile * 16 + bl;
  float* P = shP[wave];

  // Initialize state probs: P[b][s] = p0[s]
#pragma unroll
  for (int i = 0; i < 16; ++i) {
    int idx = lane * 16 + i;
    P[idx] = p0[idx & 31];
  }
  asm volatile("s_wait_dscnt 0" ::: "memory");

  const unsigned short* xb = x16 + (size_t)batch * C_SYM;
  v8f acc0, acc1;

  for (int t = 0; t < L_T; ++t) {
    // Load this step's symbol probs: 16 f16 per lane (lane b and b+16 share row b)
    union { v4u u[2]; v2h h2[8]; } X;
    const v4u* xp = (const v4u*)(xb + (size_t)t * (N_SEQ * C_SYM));
    X.u[0] = xp[0];
    X.u[1] = xp[1];
    if (t + 8 < L_T)
      __builtin_prefetch(xb + (size_t)(t + 8) * (N_SEQ * C_SYM), 0, 0);

    acc0 = (v8f){0.f, 0.f, 0.f, 0.f, 0.f, 0.f, 0.f, 0.f};
    acc1 = (v8f){0.f, 0.f, 0.f, 0.f, 0.f, 0.f, 0.f, 0.f};

#pragma unroll
    for (int kc = 0; kc < 16; ++kc) {
      // B[k,b] = p[b, 2kc+laneHi] * x[b, c] : rank-1 per column
      float ps = P[bl * 32 + 2 * kc + laneHi];
      _Float16 ph = (_Float16)ps;
      v2h pv = {ph, ph};
      union { v2h h2[8]; v16h v; } B;
#pragma unroll
      for (int j = 0; j < 8; ++j) B.h2[j] = pv * X.h2[j];  // v_pk_mul_f16

      v16h A0 = shA[kc * 32 + lane];         // states 0..15
      v16h A1 = shA[(16 + kc) * 32 + lane];  // states 16..31
      acc0 = __builtin_amdgcn_wmma_f32_16x16x32_f16(false, A0, false, B.v,
                                                    (short)0, acc0, false, false);
      acc1 = __builtin_amdgcn_wmma_f32_16x16x32_f16(false, A1, false, B.v,
                                                    (short)0, acc1, false, false);
    }

    // Write p_new back to LDS state buffer (f32, full precision between steps)
    *(v8f*)&P[bl * 32 + laneHi * 8] = acc0;        // states laneHi*8 .. +7
    *(v8f*)&P[bl * 32 + 16 + laneHi * 8] = acc1;   // states 16+laneHi*8 .. +7
    asm volatile("s_wait_dscnt 0" ::: "memory");
  }

  // log acceptance: log( sum_s (p[s]+1e-20) * sigmoid(a[s]) )
  float sum = 0.f;
#pragma unroll
  for (int j = 0; j < 8; ++j) {
    sum += (acc0[j] + 1e-20f) * accw[laneHi * 8 + j];
    sum += (acc1[j] + 1e-20f) * accw[16 + laneHi * 8 + j];
  }
  // combine lane b with lane b+16 (SWAPX16 swizzle: xor 0x10, and 0x1f)
  int oi = __builtin_amdgcn_ds_swizzle(__float_as_int(sum), 0x401F);
  float tot = sum + __int_as_float(oi);
  if (lane < 16) out[batch] = __logf(tot);
}

extern "C" void kernel_launch(void* const* d_in, const int* in_sizes, int n_in,
                              void* d_out, int out_size, void* d_ws, size_t ws_size,
                              hipStream_t stream) {
  const float* log_input = (const float*)d_in[0];   // (N,L,C)
  const float* logit_init = (const float*)d_in[1];  // (S,)
  const float* logit_T = (const float*)d_in[2];     // (S,C,S)
  const float* logit_acc = (const float*)d_in[3];   // (S,)
  float* out = (float*)d_out;                       // (1,N)

  char* ws = (char*)d_ws;
  unsigned short* wsA = (unsigned short*)(ws + WS_A);
  float* p0 = (float*)(ws + WS_P0);
  float* accw = (float*)(ws + WS_ACC);
  unsigned short* x16 = (unsigned short*)(ws + WS_X);

  // 1) exp + relayout of inputs (N*L*C = 33554432 elements)
  prep_x<<<(N_SEQ * L_T * C_SYM) / 256, 256, 0, stream>>>(log_input, x16);
  // 2) transition softmax -> WMMA A fragments
  prep_T<<<S_ST * C_SYM, 32, 0, stream>>>(logit_T, wsA);
  // 3) initial-state softmax + accepting sigmoid
  prep_misc<<<1, 32, 0, stream>>>(logit_init, logit_acc, p0, accw);
  // 4) main scan: 256 tiles of 16 batches, 4 waves/block
  pdfa_main<<<(N_SEQ / 16) / 4, 128, 0, stream>>>(x16, wsA, p0, accw, out);
}
